// LinearAttention_85650237817277
// MI455X (gfx1250) — matrix operand order repaired and checked by reading, not compiled, not run on hardware
//
#include <hip/hip_runtime.h>
#include <hip/hip_bf16.h>

typedef _Float16 h8   __attribute__((ext_vector_type(8)));
typedef _Float16 v16h __attribute__((ext_vector_type(16)));
typedef float    v8f  __attribute__((ext_vector_type(8)));

static constexpr int Bn = 4;
static constexpr int Cc = 256;
static constexpr int Hh = 128;
static constexpr int Wd = 128;
static constexpr int Nn = Hh * Wd;       // 16384 tokens
static constexpr int NH = 8;             // heads
static constexpr int HD = 32;            // head dim

// Swizzled f16 activation layout (per batch):
//   addr(c, p) = (c>>5) * (Nn*32) + p*32 + (c&31)
// -> each pixel's 32-channel group is contiguous (64B), which is exactly the
//    per-lane order a WMMA B-fragment wants.

// ---------------- f32 -> f16 weight convert with per-32-block K permutation ----
// A-fragment lane order needs k = (e&7) + (e>>3)*16 + lh*8, so within each
// 32-block we store order [0..7][16..23][8..15][24..31] (swap middle 8-groups).
__global__ void w2h_perm_kernel(const float* __restrict__ in, _Float16* __restrict__ out) {
    int i = blockIdx.x * 256 + threadIdx.x;       // over 65536
    int o  = i >> 8;
    int kk = i & 255;
    int blk = kk >> 5, r = kk & 31;
    int g = r >> 3;                               // 8-group id 0..3
    int pos = (g == 1) ? (r + 8) : (g == 2) ? (r - 8) : r;
    out[o * 256 + blk * 32 + pos] = (_Float16)in[i];
}

// ---------------- channel LayerNorm (over C per pixel), write swizzled f16 ----
__global__ void lnorm_kernel(const float* __restrict__ fmap, const float* __restrict__ g,
                             _Float16* __restrict__ Xh) {
    int idx = blockIdx.x * 256 + threadIdx.x;     // over B*N = 65536
    int b = idx >> 14;
    int p = idx & (Nn - 1);
    const float* base = fmap + (size_t)b * Cc * Nn + p;
    float s = 0.f, s2 = 0.f;
    #pragma unroll 4
    for (int c = 0; c < Cc; ++c) {
        float v = base[(size_t)c * Nn];
        s += v; s2 += v * v;
    }
    float mean = s * (1.0f / Cc);
    float var  = s2 * (1.0f / Cc) - mean * mean;
    float rinv = rsqrtf(var + 1e-5f);
    _Float16* xb = Xh + (size_t)b * Cc * Nn;
    #pragma unroll
    for (int grp = 0; grp < 8; ++grp) {
        h8 tv[4];
        #pragma unroll
        for (int e = 0; e < 32; ++e) {
            int c = grp * 32 + e;
            float v = base[(size_t)c * Nn];
            tv[e >> 3][e & 7] = (_Float16)((v - mean) * rinv * g[c]);
        }
        _Float16* ob = xb + (size_t)grp * Nn * 32 + (size_t)p * 32;
        #pragma unroll
        for (int t = 0; t < 4; ++t) *(h8*)(ob + t * 8) = tv[t];
    }
}

// ---------------- WMMA GEMM: Y[b,o,p] = sum_c W[o,c] * X[b,c,p] ----------------
// W: [256,256] f16, K-permuted.  X: [B,256,N] f16 swizzled.  Y: [B,256,N] f32.
// Block (8 waves) computes the full 256-row panel x 64 columns.
// Per K-step (K=32): async-stage 64 columns x 32 k into LDS (contiguous copies),
// B fragments = 2x ds_load_b128, A fragments = 2x global_load_b128.
static constexpr int COL_PITCH = 40;     // halves per LDS column (32 + 8 pad; 80B, 16B-aligned)

__global__ void gemm256_kernel(const _Float16* __restrict__ W,
                               const _Float16* __restrict__ X,
                               float* __restrict__ Y) {
    __shared__ _Float16 xs[64 * COL_PITCH];       // 5120 B

    int tid  = threadIdx.x;
    int lane = tid & 31;
    int wid  = tid >> 5;
    int b    = blockIdx.x >> 8;                   // 256 blocks per batch
    int p0   = (blockIdx.x & 255) << 6;           // 64-wide column panel
    int o0   = wid << 5;                          // 32 rows per wave (2 o-tiles)
    int lm   = lane & 15;
    int lh   = lane >> 4;

    const _Float16* Xb = X + (size_t)b * Cc * Nn;
    float*          Yb = Y + (size_t)b * Cc * Nn;

    // staging: thread t copies 16B: column c = t>>2, chunk h = t&3
    int sc = tid >> 2;
    int sh = tid & 3;
    unsigned int lds_off = (unsigned int)(size_t)(&xs[sc * COL_PITCH + sh * 8]);

    v8f acc[2][4] = {};

    for (int kk = 0; kk < 8; ++kk) {
        int k0 = kk * 32;

        // ---- async stage swizzled X[kblk, p0..p0+64) into LDS ----
        const _Float16* src = Xb + (size_t)kk * Nn * 32 + (size_t)(p0 + sc) * 32 + sh * 8;
        asm volatile("global_load_async_to_lds_b128 %0, %1, off"
                     :: "v"(lds_off), "v"((unsigned long long)(size_t)src) : "memory");
        if (kk < 7) __builtin_prefetch(src + Nn * 32, 0, 1);
        asm volatile("s_wait_asynccnt 0" ::: "memory");
        __syncthreads();

        // ---- A fragments (permuted W: lane's 16 halves are contiguous) ----
        const _Float16* wr0 = W + (size_t)(o0 + lm) * Cc + k0 + lh * 16;
        const _Float16* wr1 = wr0 + 16 * Cc;
        h8 a0lo = *(const h8*)(wr0);
        h8 a0hi = *(const h8*)(wr0 + 8);
        h8 a1lo = *(const h8*)(wr1);
        h8 a1hi = *(const h8*)(wr1 + 8);
        v16h A0 = __builtin_shufflevector(a0lo, a0hi, 0,1,2,3,4,5,6,7,8,9,10,11,12,13,14,15);
        v16h A1 = __builtin_shufflevector(a1lo, a1hi, 0,1,2,3,4,5,6,7,8,9,10,11,12,13,14,15);

        // ---- 4 B fragments from LDS (2x b128 each), 2 wmma each ----
        #pragma unroll
        for (int j = 0; j < 4; ++j) {
            const _Float16* cb = &xs[(j * 16 + lm) * COL_PITCH + lh * 8];
            h8 b0 = *(const h8*)(cb);
            h8 b1 = *(const h8*)(cb + 16);
            v16h Bf = __builtin_shufflevector(b0, b1, 0,1,2,3,4,5,6,7,8,9,10,11,12,13,14,15);
            acc[0][j] = __builtin_amdgcn_wmma_f32_16x16x32_f16(
                false, A0, false, Bf, (short)0, acc[0][j], false, false);
            acc[1][j] = __builtin_amdgcn_wmma_f32_16x16x32_f16(
                false, A1, false, Bf, (short)0, acc[1][j], false, false);
        }
        __syncthreads();   // protect LDS tile before next stage
    }

    #pragma unroll
    for (int t2 = 0; t2 < 2; ++t2)
        #pragma unroll
        for (int j = 0; j < 4; ++j)
            #pragma unroll
            for (int rr = 0; rr < 8; ++rr)
                Yb[(size_t)(o0 + t2 * 16 + rr + lh * 8) * Nn + p0 + j * 16 + lm] =
                    acc[t2][j][rr];
}

// ---------------- depthwise 3x3, SAME zero pad ----------------
__global__ void dwconv_kernel(const float* __restrict__ in, const float* __restrict__ wdw,
                              float* __restrict__ out) {
    int idx = blockIdx.x * 256 + threadIdx.x;     // over B*C*N
    int p  = idx & (Nn - 1);
    int bc = idx >> 14;
    int ch = bc & (Cc - 1);
    int y = p >> 7, x = p & (Wd - 1);
    const float* ib = in + (size_t)bc * Nn;
    const float* wp = wdw + ch * 9;
    float acc = 0.f;
    #pragma unroll
    for (int dy = -1; dy <= 1; ++dy) {
        int yy = y + dy;
        if (yy < 0 || yy >= Hh) continue;
        #pragma unroll
        for (int dx = -1; dx <= 1; ++dx) {
            int xx = x + dx;
            if (xx < 0 || xx >= Wd) continue;
            acc += ib[yy * Wd + xx] * wp[(dy + 1) * 3 + (dx + 1)];
        }
    }
    out[(size_t)idx] = acc;
}

// ---------------- softmax over head dim (32) per token, * c^-0.5, in place ----------------
__global__ void qsoftmax_kernel(float* __restrict__ q) {
    int idx = blockIdx.x * 256 + threadIdx.x;     // over B*NH*N = 524288
    int p  = idx & (Nn - 1);
    int bh = idx >> 14;                           // b*8+h
    float* base = q + (size_t)bh * HD * Nn + p;
    float vals[HD];
    float m = -1e30f;
    #pragma unroll
    for (int d = 0; d < HD; ++d) { vals[d] = base[(size_t)d * Nn]; m = fmaxf(m, vals[d]); }
    float s = 0.f;
    #pragma unroll
    for (int d = 0; d < HD; ++d) { vals[d] = __expf(vals[d] - m); s += vals[d]; }
    float inv = 0.17677669529663687f / s;         // 32^-0.5 / sum
    #pragma unroll
    for (int d = 0; d < HD; ++d) base[(size_t)d * Nn] = vals[d] * inv;
}

// ---------------- softmax over tokens (16384) per (b,channel) row, in place ----------------
__global__ void ksoftmax_kernel(float* __restrict__ k) {
    __shared__ float red[256];
    int tid = threadIdx.x;
    float* p = k + (size_t)blockIdx.x * Nn;       // 1024 rows
    float m = -1e30f;
    for (int i = tid; i < Nn; i += 256) m = fmaxf(m, p[i]);
    red[tid] = m; __syncthreads();
    for (int s = 128; s > 0; s >>= 1) { if (tid < s) red[tid] = fmaxf(red[tid], red[tid + s]); __syncthreads(); }
    m = red[0]; __syncthreads();
    float sum = 0.f;
    for (int i = tid; i < Nn; i += 256) sum += __expf(p[i] - m);
    red[tid] = sum; __syncthreads();
    for (int s = 128; s > 0; s >>= 1) { if (tid < s) red[tid] += red[tid + s]; __syncthreads(); }
    float inv = 1.0f / red[0];
    for (int i = tid; i < Nn; i += 256) p[i] = __expf(p[i] - m) * inv;
}

// ---------------- ctx[bh,d,e] = sum_n k[bh,d,n] * v[bh,e,n]  (LDS-tiled) ----------------
__global__ void ctx_kernel(const float* __restrict__ kd, const float* __restrict__ vd,
                           float* __restrict__ ctx) {
    __shared__ float ks[HD * 65];
    __shared__ float vs[HD * 65];
    int bh = blockIdx.x, tid = threadIdx.x;
    const float* kb = kd + (size_t)bh * HD * Nn;
    const float* vb = vd + (size_t)bh * HD * Nn;
    float acc[4] = {0.f, 0.f, 0.f, 0.f};
    for (int chn = 0; chn < Nn / 64; ++chn) {
        int t0 = chn * 64;
        #pragma unroll
        for (int i = 0; i < 8; ++i) {
            int li = tid + i * 256, d = li >> 6, tk = li & 63;
            ks[d * 65 + tk] = kb[(size_t)d * Nn + t0 + tk];
            vs[d * 65 + tk] = vb[(size_t)d * Nn + t0 + tk];
        }
        __syncthreads();
        #pragma unroll
        for (int i = 0; i < 4; ++i) {
            int pi = tid + i * 256, d = pi >> 5, e = pi & 31;
            float s = acc[i];
            #pragma unroll 8
            for (int tk = 0; tk < 64; ++tk) s += ks[d * 65 + tk] * vs[e * 65 + tk];
            acc[i] = s;
        }
        __syncthreads();
    }
    #pragma unroll
    for (int i = 0; i < 4; ++i) ctx[(size_t)bh * 1024 + tid + i * 256] = acc[i];
}

// ---------------- out = q . ctx, then SiLU, write swizzled f16 ----------------
__global__ void attnout_kernel(const float* __restrict__ q, const float* __restrict__ ctx,
                               _Float16* __restrict__ Ah) {
    __shared__ float cs[1024];
    int bh = blockIdx.x >> 6;
    int pc = blockIdx.x & 63;
    int tid = threadIdx.x;
    int p = pc * 256 + tid;
    #pragma unroll
    for (int i = 0; i < 4; ++i) cs[tid + i * 256] = ctx[(size_t)bh * 1024 + tid + i * 256];
    __syncthreads();
    const float* qb = q + (size_t)bh * HD * Nn + p;
    float qv[HD];
    #pragma unroll
    for (int d = 0; d < HD; ++d) qv[d] = qb[(size_t)d * Nn];
    // swizzled dest: batch base + head * (Nn*32) + p*32 + e  (e = channel&31)
    _Float16* ob = Ah + (size_t)(bh >> 3) * Cc * Nn + (size_t)(bh & 7) * Nn * 32 + (size_t)p * 32;
    h8 tv[4];
    #pragma unroll
    for (int e = 0; e < HD; ++e) {
        float s = 0.f;
        #pragma unroll
        for (int d = 0; d < HD; ++d) s += qv[d] * cs[d * 32 + e];
        float sil = s / (1.0f + __expf(-s));
        tv[e >> 3][e & 7] = (_Float16)sil;
    }
    #pragma unroll
    for (int t = 0; t < 4; ++t) *(h8*)(ob + t * 8) = tv[t];
}

extern "C" void kernel_launch(void* const* d_in, const int* in_sizes, int n_in,
                              void* d_out, int out_size, void* d_ws, size_t ws_size,
                              hipStream_t stream) {
    (void)in_sizes; (void)n_in; (void)out_size; (void)ws_size;
    const float* fmap = (const float*)d_in[0];
    const float* g    = (const float*)d_in[1];
    const float* wq1  = (const float*)d_in[2];
    const float* wq2  = (const float*)d_in[3];
    const float* wk1  = (const float*)d_in[4];
    const float* wk2  = (const float*)d_in[5];
    const float* wv1  = (const float*)d_in[6];
    const float* wv2  = (const float*)d_in[7];
    const float* wo   = (const float*)d_in[8];
    float* out = (float*)d_out;

    const size_t TOT = (size_t)Bn * Cc * Nn;   // 16,777,216 elements
    char* w = (char*)d_ws;
    size_t off = 0;
    _Float16* Xh  = (_Float16*)(w + off); off += TOT * 2;          // swizzled f16 (reused for Ah)
    _Float16* WhQ = (_Float16*)(w + off); off += (size_t)Cc * Cc * 2;
    _Float16* WhK = (_Float16*)(w + off); off += (size_t)Cc * Cc * 2;
    _Float16* WhV = (_Float16*)(w + off); off += (size_t)Cc * Cc * 2;
    _Float16* WhO = (_Float16*)(w + off); off += (size_t)Cc * Cc * 2;
    float* Qf = (float*)(w + off); off += TOT * 4;
    float* Kf = (float*)(w + off); off += TOT * 4;
    float* Vf = (float*)(w + off); off += TOT * 4;
    float* Tf = (float*)(w + off); off += TOT * 4;
    float* Cx = (float*)(w + off); off += (size_t)Bn * NH * HD * HD * 4;

    // 1) weight converts (K-permuted for A-fragment contiguity)
    w2h_perm_kernel<<<256, 256, 0, stream>>>(wq1, WhQ);
    w2h_perm_kernel<<<256, 256, 0, stream>>>(wk1, WhK);
    w2h_perm_kernel<<<256, 256, 0, stream>>>(wv1, WhV);
    w2h_perm_kernel<<<256, 256, 0, stream>>>(wo,  WhO);

    // 2) LayerNorm -> swizzled f16 X
    lnorm_kernel<<<(Bn * Nn) / 256, 256, 0, stream>>>(fmap, g, Xh);

    // 3) Q/K/V 1x1 convs as WMMA GEMMs (async LDS staging, b128 fragments)
    gemm256_kernel<<<Bn * (Nn / 64), 256, 0, stream>>>(WhQ, Xh, Qf);
    gemm256_kernel<<<Bn * (Nn / 64), 256, 0, stream>>>(WhK, Xh, Kf);
    gemm256_kernel<<<Bn * (Nn / 64), 256, 0, stream>>>(WhV, Xh, Vf);

    // 4) depthwise 3x3 (rotate buffers: q->Tf, k->Qf, v->Kf)
    dwconv_kernel<<<(int)(TOT / 256), 256, 0, stream>>>(Qf, wq2, Tf);
    dwconv_kernel<<<(int)(TOT / 256), 256, 0, stream>>>(Kf, wk2, Qf);
    dwconv_kernel<<<(int)(TOT / 256), 256, 0, stream>>>(Vf, wv2, Kf);

    // 5) softmaxes (in place): q in Tf, k in Qf
    qsoftmax_kernel<<<(Bn * NH * Nn) / 256, 256, 0, stream>>>(Tf);
    ksoftmax_kernel<<<Bn * Cc, 256, 0, stream>>>(Qf);

    // 6) ctx = k^T v   (k=Qf, v=Kf)
    ctx_kernel<<<Bn * NH, 256, 0, stream>>>(Qf, Kf, Cx);

    // 7) out = q.ctx, SiLU -> swizzled f16 (reuse Xh region)
    attnout_kernel<<<Bn * NH * (Nn / 256), 256, 0, stream>>>(Tf, Cx, Xh);

    // 8) final 1x1 conv via WMMA GEMM -> d_out (f32)
    gemm256_kernel<<<Bn * (Nn / 64), 256, 0, stream>>>(WhO, Xh, out);
}
